// _Head_S1_37623913513354
// MI455X (gfx1250) — matrix operand order repaired and checked by reading, not compiled, not run on hardware
//
#include <hip/hip_runtime.h>
#include <hip/hip_bf16.h>

// ---------------------------------------------------------------------------
// Types for CDNA5 WMMA (wave32): D = A(16x32 bf16) x B(32x16 bf16) + C(16x16 f32)
// ---------------------------------------------------------------------------
typedef __attribute__((ext_vector_type(16))) __bf16 bf16x16;
typedef __attribute__((ext_vector_type(8)))  float  f32x8;

union FragBF {
    bf16x16 v;
    uint4   q[2];
};

__device__ __forceinline__ unsigned short f32_to_bf16_rne(float f) {
    unsigned u = __float_as_uint(f);
    unsigned r = u + 0x7FFFu + ((u >> 16) & 1u);
    return (unsigned short)(r >> 16);
}

// ---------------------------------------------------------------------------
// Conversion / padding kernels
// ---------------------------------------------------------------------------
// x [128,49,512] f32  ->  xb [128,64,512] bf16, rows s>=49 zero-padded.
__global__ __launch_bounds__(256) void k_pad_convert_x(const float* __restrict__ x,
                                                       unsigned short* __restrict__ xb) {
    int idx = blockIdx.x * 256 + threadIdx.x;      // 128*64*512 = 4194304 exact
    int c = idx & 511;
    int r = (idx >> 9) & 63;
    int b = idx >> 15;
    float v = (r < 49) ? x[((size_t)b * 49 + r) * 512 + c] : 0.0f;
    xb[idx] = f32_to_bf16_rne(v);
}

__global__ __launch_bounds__(256) void k_f32_to_bf16(const float* __restrict__ in,
                                                     unsigned short* __restrict__ out, int n) {
    int idx = blockIdx.x * 256 + threadIdx.x;
    if (idx < n) out[idx] = f32_to_bf16_rne(in[idx]);
}

// ---------------------------------------------------------------------------
// Core: one wave computes a 64x64 f32 tile of A(64x512) x B(512x64)^T.
// A row-major [M,512] bf16 ; B rows are the N dimension: B[n, k] (row-major,
// K contiguous) -- exactly the layout of lin_W[k][d, c], U_W, V_W.
// Per ISA layouts (CDNA5 ISA):
//   A frag : lane(lm,lh): halfs  A[(16i+lm)*512 + kk + 8*lh  + 0..7]
//                          and   A[(16i+lm)*512 + kk + 16+8*lh + 0..7]
//   B frag : lane(lm,lh): halfs  B[(16j+lm)*512 + kk + 8*lh  + 0..7]
//                          and   B[(16j+lm)*512 + kk + 16+8*lh + 0..7]
//   C/D    : elem v -> M = 16i + v + 8*lh, N = 16j + lm
// ---------------------------------------------------------------------------
__device__ __forceinline__ void wmma_tile_64x64(const __bf16* __restrict__ A,
                                                const __bf16* __restrict__ B,
                                                int lm, int lh, f32x8 acc[4][4]) {
#pragma unroll 1
    for (int kk = 0; kk < 512; kk += 32) {
        FragBF a[4], bm[4];
#pragma unroll
        for (int i = 0; i < 4; ++i) {
            const __bf16* p = A + (size_t)(i * 16 + lm) * 512 + kk + lh * 8;
            a[i].q[0] = *reinterpret_cast<const uint4*>(p);
            a[i].q[1] = *reinterpret_cast<const uint4*>(p + 16);
        }
#pragma unroll
        for (int j = 0; j < 4; ++j) {
            const __bf16* p = B + (size_t)(j * 16 + lm) * 512 + kk + lh * 8;
            bm[j].q[0] = *reinterpret_cast<const uint4*>(p);
            bm[j].q[1] = *reinterpret_cast<const uint4*>(p + 16);
        }
#pragma unroll
        for (int i = 0; i < 4; ++i)
#pragma unroll
            for (int j = 0; j < 4; ++j)
                acc[i][j] = __builtin_amdgcn_wmma_f32_16x16x32_bf16(
                    false, a[i].v, false, bm[j].v, (short)0, acc[i][j], false, false);
    }
}

// ---------------------------------------------------------------------------
// Pass 1: g = x @ W[k]^T (no bias; padded rows give exact 0) ; accumulate
// per-(class,feature) sum / sumsq over (b,s) via atomics.  grid = 27*128.
// ---------------------------------------------------------------------------
__global__ __launch_bounds__(256) void k_gemm_stats(const __bf16* __restrict__ xb,
                                                    const __bf16* __restrict__ Wb,
                                                    float* __restrict__ sum1,
                                                    float* __restrict__ sumsq1) {
    int kc = blockIdx.x / 128;
    int b  = blockIdx.x % 128;
    int wave = threadIdx.x >> 5;
    int lane = threadIdx.x & 31;
    int lm = lane & 15, lh = lane >> 4;
    const __bf16* A = xb + (size_t)b * 64 * 512;
    const __bf16* B = Wb + (size_t)kc * 512 * 512 + (size_t)wave * 64 * 512;
    f32x8 acc[4][4];
    const f32x8 z = {0.f, 0.f, 0.f, 0.f, 0.f, 0.f, 0.f, 0.f};
#pragma unroll
    for (int i = 0; i < 4; ++i)
#pragma unroll
        for (int j = 0; j < 4; ++j) acc[i][j] = z;

    wmma_tile_64x64(A, B, lm, lh, acc);

#pragma unroll
    for (int j = 0; j < 4; ++j) {
        float s1 = 0.f, s2 = 0.f;
#pragma unroll
        for (int i = 0; i < 4; ++i)
#pragma unroll
            for (int v = 0; v < 8; ++v) {
                float g = acc[i][j][v];
                s1 += g;
                s2 += g * g;
            }
        s1 += __shfl_xor(s1, 16, 32);   // combine the two row-halves of each column
        s2 += __shfl_xor(s2, 16, 32);
        if (lane < 16) {
            int col = kc * 512 + wave * 64 + j * 16 + lane;
            atomicAdd(&sum1[col], s1);
            atomicAdd(&sumsq1[col], s2);
        }
    }
}

// Fold BN:  f_norm = (g - mean_g) * rsqrt(var+eps) * gamma + beta
// (linear bias cancels inside BN).  sA = gamma*rs ; sB = beta - mean_g*sA.
__global__ __launch_bounds__(256) void k_bn1_finalize(const float* __restrict__ sum1,
                                                      const float* __restrict__ sumsq1,
                                                      const float* __restrict__ gamma,
                                                      const float* __restrict__ beta,
                                                      float* __restrict__ sA,
                                                      float* __restrict__ sB) {
    int idx = blockIdx.x * 256 + threadIdx.x;   // 27*512 = 13824
    if (idx >= 27 * 512) return;
    const float invM = 1.0f / 6272.0f;          // b*s = 128*49 valid rows
    float m   = sum1[idx] * invM;
    float var = fmaxf(sumsq1[idx] * invM - m * m, 0.0f);
    float rs  = rsqrtf(var + 1e-5f);
    float sc  = gamma[idx] * rs;
    sA[idx] = sc;
    sB[idx] = beta[idx] - m * sc;
}

// ---------------------------------------------------------------------------
// Pass 2: recompute g, apply BN+ReLU, mean over s (mask padded rows),
// write f_v[b,27,512].  grid = 27*128.
// ---------------------------------------------------------------------------
__global__ __launch_bounds__(256) void k_gemm_bn_relu_mean(const __bf16* __restrict__ xb,
                                                           const __bf16* __restrict__ Wb,
                                                           const float* __restrict__ sA,
                                                           const float* __restrict__ sB,
                                                           float* __restrict__ fv) {
    int kc = blockIdx.x / 128;
    int b  = blockIdx.x % 128;
    int wave = threadIdx.x >> 5;
    int lane = threadIdx.x & 31;
    int lm = lane & 15, lh = lane >> 4;
    const __bf16* A = xb + (size_t)b * 64 * 512;
    const __bf16* B = Wb + (size_t)kc * 512 * 512 + (size_t)wave * 64 * 512;
    f32x8 acc[4][4];
    const f32x8 z = {0.f, 0.f, 0.f, 0.f, 0.f, 0.f, 0.f, 0.f};
#pragma unroll
    for (int i = 0; i < 4; ++i)
#pragma unroll
        for (int j = 0; j < 4; ++j) acc[i][j] = z;

    wmma_tile_64x64(A, B, lm, lh, acc);

#pragma unroll
    for (int j = 0; j < 4; ++j) {
        int col = kc * 512 + wave * 64 + j * 16 + lm;
        float a = sA[col], sh = sB[col];
        float s = 0.f;
#pragma unroll
        for (int i = 0; i < 4; ++i)
#pragma unroll
            for (int v = 0; v < 8; ++v) {
                int srow = i * 16 + v + lh * 8;           // s index within batch
                float h = fmaxf(acc[i][j][v] * a + sh, 0.0f);
                s += (srow < 49) ? h : 0.0f;
            }
        s += __shfl_xor(s, 16, 32);
        if (lane < 16)
            fv[((size_t)b * 27 + kc) * 512 + wave * 64 + j * 16 + lane] = s * (1.0f / 49.0f);
    }
}

// ---------------------------------------------------------------------------
// Generic C = A @ W^T + bias, M = 3456 (=54*64).  grid = 54.
// ---------------------------------------------------------------------------
__global__ __launch_bounds__(256) void k_gemm_bias(const __bf16* __restrict__ Ab,
                                                   const __bf16* __restrict__ Wb,
                                                   const float* __restrict__ bias,
                                                   float* __restrict__ out) {
    int mbase = blockIdx.x * 64;
    int wave = threadIdx.x >> 5;
    int lane = threadIdx.x & 31;
    int lm = lane & 15, lh = lane >> 4;
    const __bf16* A = Ab + (size_t)mbase * 512;
    const __bf16* B = Wb + (size_t)wave * 64 * 512;
    f32x8 acc[4][4];
    const f32x8 z = {0.f, 0.f, 0.f, 0.f, 0.f, 0.f, 0.f, 0.f};
#pragma unroll
    for (int i = 0; i < 4; ++i)
#pragma unroll
        for (int j = 0; j < 4; ++j) acc[i][j] = z;

    wmma_tile_64x64(A, B, lm, lh, acc);

#pragma unroll
    for (int j = 0; j < 4; ++j) {
        int col = wave * 64 + j * 16 + lm;
        float bv = bias[col];
#pragma unroll
        for (int i = 0; i < 4; ++i)
#pragma unroll
            for (int v = 0; v < 8; ++v) {
                int row = mbase + i * 16 + v + lh * 8;
                out[(size_t)row * 512 + col] = acc[i][j][v] + bv;
            }
    }
}

// ---------------------------------------------------------------------------
// Per-batch graph: si = fv fv^T, top-4 threshold, D^-1/2 A D^-1/2,
// agg = A @ Vx + Ux ; accumulate per-node sum/sumsq for BN.  grid = 128.
// ---------------------------------------------------------------------------
__global__ __launch_bounds__(256) void k_graph(const float* __restrict__ fv,
                                               const float* __restrict__ Vx,
                                               const float* __restrict__ Ux,
                                               float* __restrict__ agg,
                                               float* __restrict__ sum2,
                                               float* __restrict__ sumsq2) {
    __shared__ float fs[27 * 512];
    __shared__ float si[27 * 27];
    __shared__ float thr[27];
    __shared__ float degis[27];
    __shared__ float ls[27], lq[27];

    int b = blockIdx.x, tid = threadIdx.x;
    const float* fb = fv + (size_t)b * 27 * 512;
    for (int i = tid; i < 27 * 512; i += 256) fs[i] = fb[i];
    if (tid < 27) { ls[tid] = 0.f; lq[tid] = 0.f; }
    __syncthreads();

    for (int p = tid; p < 729; p += 256) {
        int i = p / 27, j = p % 27;
        float s = 0.f;
        for (int c = 0; c < 512; ++c) s += fs[i * 512 + c] * fs[j * 512 + c];
        si[p] = s;
    }
    __syncthreads();

    if (tid < 27) {
        unsigned mask = 0;
        float t = 0.f;
        for (int p = 0; p < 4; ++p) {             // 4th largest (kth order stat)
            float m = -1e30f;
            int am = 0;
            for (int j = 0; j < 27; ++j) {
                float v = si[tid * 27 + j];
                if (!((mask >> j) & 1u) && v > m) { m = v; am = j; }
            }
            mask |= 1u << am;
            t = m;
        }
        thr[tid] = t;
        int deg = 0;
        for (int j = 0; j < 27; ++j) deg += (si[tid * 27 + j] >= t) ? 1 : 0;
        degis[tid] = rsqrtf((float)deg);
    }
    __syncthreads();

    const float* Vb = Vx + (size_t)b * 27 * 512;
    const float* Ub = Ux + (size_t)b * 27 * 512;
    for (int p = tid; p < 27 * 512; p += 256) {
        int i = p >> 9, c = p & 511;
        float a = Ub[p];
        float di = degis[i], ti = thr[i];
        for (int j = 0; j < 27; ++j) {
            float w = (si[i * 27 + j] >= ti) ? di * degis[j] : 0.0f;
            a += w * Vb[(size_t)j * 512 + c];
        }
        agg[(size_t)b * 27 * 512 + p] = a;
        atomicAdd(&ls[i], a);
        atomicAdd(&lq[i], a * a);
    }
    __syncthreads();
    if (tid < 27) {
        atomicAdd(&sum2[tid], ls[tid]);
        atomicAdd(&sumsq2[tid], lq[tid]);
    }
}

__global__ void k_bn2_finalize(const float* __restrict__ sum2, const float* __restrict__ sumsq2,
                               const float* __restrict__ g, const float* __restrict__ be,
                               float* __restrict__ scale2, float* __restrict__ shift2) {
    int i = threadIdx.x;
    if (i >= 27) return;
    const float invC = 1.0f / 65536.0f;           // 128 * 512
    float mu  = sum2[i] * invC;
    float var = fmaxf(sumsq2[i] * invC - mu * mu, 0.0f);
    float rs  = rsqrtf(var + 1e-5f);
    float sc  = g[i] * rs;
    scale2[i] = sc;
    shift2[i] = be[i] - mu * sc;
}

// L2-normalized ReLU prototypes: rows 0..26 = main_sc, 27..40 = sub_sc. grid=41.
__global__ __launch_bounds__(256) void k_proto(const float* __restrict__ main_sc,
                                               const float* __restrict__ sub_sc,
                                               float* __restrict__ main_n,
                                               float* __restrict__ sub_n) {
    int r = blockIdx.x, tid = threadIdx.x;
    const float* src = (r < 27) ? main_sc + (size_t)r * 512 : sub_sc + (size_t)(r - 27) * 512;
    float* dst       = (r < 27) ? main_n + (size_t)r * 512 : sub_n + (size_t)(r - 27) * 512;
    __shared__ float red[256];
    float s = 0.f;
    for (int c = tid; c < 512; c += 256) {
        float v = fmaxf(src[c], 0.0f);
        s += v * v;
    }
    red[tid] = s;
    __syncthreads();
    for (int o = 128; o > 0; o >>= 1) {
        if (tid < o) red[tid] += red[tid + o];
        __syncthreads();
    }
    float inv = 1.0f / fmaxf(sqrtf(red[0]), 1e-12f);
    for (int c = tid; c < 512; c += 256) dst[c] = fmaxf(src[c], 0.0f) * inv;
}

// SUB_LIST = (0,1,2,4,7,8,11) -> position or -1
__constant__ int c_subpos[27] = {0, 1, 2, -1, 3, -1, -1, 4, 5, -1, -1, 6,
                                 -1, -1, -1, -1, -1, -1, -1, -1, -1, -1, -1, -1, -1, -1, -1};

// Residual+BN+ReLU then cosine head.  grid = 128*27 (one block per (b,i)).
__global__ __launch_bounds__(256) void k_final(const float* __restrict__ fv,
                                               const float* __restrict__ agg,
                                               const float* __restrict__ scale2,
                                               const float* __restrict__ shift2,
                                               const float* __restrict__ main_n,
                                               const float* __restrict__ sub_n,
                                               float* __restrict__ out) {
    int blk = blockIdx.x;
    int b = blk / 27, i = blk % 27;
    int tid = threadIdx.x;
    float s2 = scale2[i], sh = shift2[i];
    const float* fr = fv + ((size_t)b * 27 + i) * 512;
    const float* ar = agg + ((size_t)b * 27 + i) * 512;
    const float* mn = main_n + (size_t)i * 512;
    int p = c_subpos[i];
    const float* p0 = (p >= 0) ? sub_n + (size_t)(2 * p) * 512 : mn;
    const float* p1 = (p >= 0) ? sub_n + (size_t)(2 * p + 1) * 512 : mn;

    float nrm = 0.f, dm = 0.f, d0 = 0.f, d1 = 0.f;
    for (int c = tid; c < 512; c += 256) {
        float v = fmaxf(fr[c] + ar[c] * s2 + sh, 0.0f);
        nrm += v * v;
        dm += v * mn[c];
        d0 += v * p0[c];
        d1 += v * p1[c];
    }
    __shared__ float r0[256], r1[256], r2[256], r3[256];
    r0[tid] = nrm; r1[tid] = dm; r2[tid] = d0; r3[tid] = d1;
    __syncthreads();
    for (int o = 128; o > 0; o >>= 1) {
        if (tid < o) {
            r0[tid] += r0[tid + o];
            r1[tid] += r1[tid + o];
            r2[tid] += r2[tid + o];
            r3[tid] += r3[tid + o];
        }
        __syncthreads();
    }
    if (tid == 0) {
        float inv = 1.0f / fmaxf(sqrtf(r0[0]), 1e-12f);
        out[b * 41 + i] = r1[0] * inv;
        if (p >= 0) {
            out[b * 41 + 27 + 2 * p]     = r2[0] * inv;
            out[b * 41 + 27 + 2 * p + 1] = r3[0] * inv;
        }
    }
}

// ---------------------------------------------------------------------------
// Host-side launcher
// ---------------------------------------------------------------------------
extern "C" void kernel_launch(void* const* d_in, const int* in_sizes, int n_in,
                              void* d_out, int out_size, void* d_ws, size_t ws_size,
                              hipStream_t stream) {
    (void)in_sizes; (void)n_in; (void)out_size; (void)ws_size;

    const float* x         = (const float*)d_in[0];   // [128,49,512]
    const float* lin_W     = (const float*)d_in[1];   // [27,512,512]
    // d_in[2] = lin_b: cancels inside BatchNorm, unused.
    const float* lin_gamma = (const float*)d_in[3];   // [27,512]
    const float* lin_beta  = (const float*)d_in[4];   // [27,512]
    const float* U_W       = (const float*)d_in[5];   // [512,512]
    const float* U_b       = (const float*)d_in[6];   // [512]
    const float* V_W       = (const float*)d_in[7];   // [512,512]
    const float* V_b       = (const float*)d_in[8];   // [512]
    const float* bnv_gamma = (const float*)d_in[9];   // [27]
    const float* bnv_beta  = (const float*)d_in[10];  // [27]
    const float* main_sc   = (const float*)d_in[11];  // [27,512]
    const float* sub_sc    = (const float*)d_in[12];  // [14,512]
    float* out = (float*)d_out;                       // [128,41]

    // Workspace carve (each chunk 256B aligned; total ~56 MB)
    char* w = (char*)d_ws;
    size_t off = 0;
    auto carve = [&](size_t bytes) {
        char* p = w + off;
        off += (bytes + 255) & ~(size_t)255;
        return p;
    };
    unsigned short* xb    = (unsigned short*)carve((size_t)128 * 64 * 512 * 2);
    unsigned short* linWb = (unsigned short*)carve((size_t)27 * 512 * 512 * 2);
    unsigned short* UWb   = (unsigned short*)carve((size_t)512 * 512 * 2);
    unsigned short* VWb   = (unsigned short*)carve((size_t)512 * 512 * 2);
    unsigned short* fvb   = (unsigned short*)carve((size_t)3456 * 512 * 2);
    float* fv     = (float*)carve((size_t)3456 * 512 * 4);
    float* Vx     = (float*)carve((size_t)3456 * 512 * 4);
    float* Ux     = (float*)carve((size_t)3456 * 512 * 4);
    float* agg    = (float*)carve((size_t)3456 * 512 * 4);
    float* sum1   = (float*)carve((size_t)27 * 512 * 4);   // contiguous zero region start
    float* sumsq1 = (float*)carve((size_t)27 * 512 * 4);
    float* sum2   = (float*)carve(256);
    float* sumsq2 = (float*)carve(256);                    // contiguous zero region end
    float* sA     = (float*)carve((size_t)27 * 512 * 4);
    float* sB     = (float*)carve((size_t)27 * 512 * 4);
    float* scale2 = (float*)carve(256);
    float* shift2 = (float*)carve(256);
    float* main_n = (float*)carve((size_t)27 * 512 * 4);
    float* sub_n  = (float*)carve((size_t)14 * 512 * 4);

    // Zero the atomic accumulators (55296+55296+256+256 bytes, contiguous).
    hipMemsetAsync(sum1, 0, 111104, stream);

    // 1) convert inputs to bf16 (x padded s:49->64 with zeros)
    k_pad_convert_x<<<16384, 256, 0, stream>>>(x, xb);
    k_f32_to_bf16<<<27648, 256, 0, stream>>>(lin_W, linWb, 27 * 512 * 512);
    k_f32_to_bf16<<<1024, 256, 0, stream>>>(U_W, UWb, 512 * 512);
    k_f32_to_bf16<<<1024, 256, 0, stream>>>(V_W, VWb, 512 * 512);

    // 2) GEMM pass 1: BN statistics
    k_gemm_stats<<<27 * 128, 256, 0, stream>>>((const __bf16*)xb, (const __bf16*)linWb,
                                               sum1, sumsq1);
    k_bn1_finalize<<<54, 256, 0, stream>>>(sum1, sumsq1, lin_gamma, lin_beta, sA, sB);

    // 3) GEMM pass 2: BN + ReLU + mean over s  -> f_v [128,27,512]
    k_gemm_bn_relu_mean<<<27 * 128, 256, 0, stream>>>((const __bf16*)xb, (const __bf16*)linWb,
                                                      sA, sB, fv);

    // 4) Vx / Ux GEMMs
    k_f32_to_bf16<<<6912, 256, 0, stream>>>(fv, fvb, 3456 * 512);
    k_gemm_bias<<<54, 256, 0, stream>>>((const __bf16*)fvb, (const __bf16*)VWb, V_b, Vx);
    k_gemm_bias<<<54, 256, 0, stream>>>((const __bf16*)fvb, (const __bf16*)UWb, U_b, Ux);

    // 5) dynamic top-4 graph + aggregation + node-BN stats
    k_graph<<<128, 256, 0, stream>>>(fv, Vx, Ux, agg, sum2, sumsq2);
    k_bn2_finalize<<<1, 32, 0, stream>>>(sum2, sumsq2, bnv_gamma, bnv_beta, scale2, shift2);

    // 6) prototypes + cosine head
    k_proto<<<41, 256, 0, stream>>>(main_sc, sub_sc, main_n, sub_n);
    k_final<<<128 * 27, 256, 0, stream>>>(fv, agg, scale2, shift2, main_n, sub_n, out);
}
